// GRULoop_60026462929029
// MI455X (gfx1250) — matrix operand order repaired and checked by reading, not compiled, not run on hardware
//
#include <hip/hip_runtime.h>
#include <cstdint>

typedef __attribute__((ext_vector_type(16))) __bf16 v16bf;
typedef __attribute__((ext_vector_type(8)))  __bf16 v8bf;
typedef __attribute__((ext_vector_type(8)))  float  v8f;
typedef __attribute__((ext_vector_type(4)))  unsigned int v4u;
typedef __attribute__((ext_vector_type(8)))  int v8i;
typedef __attribute__((ext_vector_type(4)))  int v4i;

#define UNITS  8
#define BATCH  65536
#define IDIM   256
#define HDIM   256
#define G3     768

#define BM     64          // batch rows per block
#define NW     4           // waves per block (wave32)
#define SP     264         // LDS row stride (bf16): 132 dwords == 4 mod 64 banks -> conflict-free
#define NSLICE (UNITS * 16)

#if defined(__HIP_DEVICE_COMPILE__) && __has_builtin(__builtin_amdgcn_tensor_load_to_lds)
#define HAS_TDM 1
#else
#define HAS_TDM 0
#endif

union V16U { v16bf v; v8bf h[2]; };

// sigmoid/tanh with v_exp_f32 + v_rcp_f32 (no IEEE divide expansion)
__device__ __forceinline__ float fsigmoid(float x) {
    return __builtin_amdgcn_rcpf(1.0f + __expf(-x));
}
__device__ __forceinline__ float ftanh_fast(float x) {
    float e = __expf(2.0f * x);
    return (e - 1.0f) * __builtin_amdgcn_rcpf(e + 1.0f);
}

// f32 -> bf16 weight conversion (runs once; weights then live hot in L2)
__global__ void cvt_bf16_kernel(const float* __restrict__ src,
                                __bf16* __restrict__ dst, int n) {
    int i = blockIdx.x * blockDim.x + threadIdx.x;
    if (i < n) dst[i] = (__bf16)src[i];
}

#if HAS_TDM
// Issue 6 TDM loads (3 gates x {Wih,Whh}) for weight slice `sc` (= u*16 + jt):
// each is a 16-row x 256-elem bf16 tile, DMA'd into LDS with pad_enable adding
// 4 dwords per 128-dword row -> LDS row stride of 264 bf16 elements.
__device__ __forceinline__ void tdm_issue_slice(const __bf16* __restrict__ wih_bf,
                                                const __bf16* __restrict__ whh_bf,
                                                int sc, __bf16* dst)
{
    const int uu = sc >> 4;
    const int jb = (sc & 15) * 16;
    #pragma unroll
    for (int t = 0; t < 6; ++t) {
        const __bf16* mat = (t < 3) ? wih_bf : whh_bf;
        const int gate = (t < 3) ? t : (t - 3);
        const uint64_t ga = (uint64_t)(uintptr_t)(mat + ((size_t)uu * G3 + gate * 256 + jb) * 256);
        const uint32_t la = (uint32_t)(uintptr_t)(dst + t * 16 * SP);

        v4u g0;
        g0.x = 1u;                                   // count=1 (valid), gather off
        g0.y = la;                                   // lds_addr
        g0.z = (uint32_t)ga;                         // global_addr[31:0]
        g0.w = ((uint32_t)(ga >> 32) & 0x01FFFFFFu)  // global_addr[56:32]
             | (2u << 30);                           // type = 2 ("image")

        v8i g1;
        g1[0] = (int)((1u << 16)     // data_size = 2 bytes
                    | (1u << 20)     // pad_enable
                    | (6u << 22)     // pad_interval: 128 dwords (one 256-elem row)
                    | (3u << 25));   // pad_amount: 4 dwords (8 bf16) -> stride 264
        g1[1] = (int)(256u << 16);   // tensor_dim0 = 256  (bits 79:48, low half)
        g1[2] = (int)(16u  << 16);   // tensor_dim0 hi | tensor_dim1 = 16
        g1[3] = (int)(256u << 16);   // tensor_dim1 hi | tile_dim0 = 256
        g1[4] = 16;                  // tile_dim1 = 16, tile_dim2 = 0
        g1[5] = 256;                 // tensor_dim0_stride = 256 elems
        g1[6] = 0;
        g1[7] = 0;

        const v4i z4 = {0, 0, 0, 0};
#if __clang_major__ >= 23
        const v8i z8 = {0, 0, 0, 0, 0, 0, 0, 0};
        __builtin_amdgcn_tensor_load_to_lds(g0, g1, z4, z4, z8, 0);
#else
        __builtin_amdgcn_tensor_load_to_lds(g0, g1, z4, z4, 0);
#endif
    }
}
#endif

// Fallback: cooperative copy of the weight slice into LDS (all threads).
__device__ __forceinline__ void copy_slice(const __bf16* __restrict__ wih_bf,
                                           const __bf16* __restrict__ whh_bf,
                                           int sc, __bf16* dst, int tid)
{
    const int uu = sc >> 4;
    const int jb = (sc & 15) * 16;
    for (int idx = tid * 8; idx < 96 * 256; idx += NW * 32 * 8) {
        const int r = idx >> 8;                    // 0..95
        const int c = idx & 255;
        const __bf16* mat = (r < 48) ? wih_bf : whh_bf;
        const int gate = (r >> 4) % 3;
        const size_t srow = (size_t)uu * G3 + gate * 256 + jb + (r & 15);
        *(v8bf*)&dst[r * SP + c] = *(const v8bf*)&mat[srow * 256 + c];
    }
}

__global__ __launch_bounds__(NW * 32, 1)
void gru_chain_kernel(const float* __restrict__ x,          // [B][256] f32
                      const __bf16* __restrict__ wih_bf,    // [8][768][256] bf16
                      const __bf16* __restrict__ whh_bf,    // [8][768][256] bf16
                      const float* __restrict__ b_ih,       // [8][768]
                      const float* __restrict__ b_hh,       // [8][768]
                      const float* __restrict__ h0,         // [256]
                      float* __restrict__ out)              // [B][256]
{
    __shared__ __bf16 hsA[BM * SP];
    __shared__ __bf16 hsB[BM * SP];
    __shared__ __bf16 wsl[2][96 * SP];   // double-buffered weight slice (TDM target)

    const int tid  = threadIdx.x;
    const int wave = tid >> 5;
    const int lane = tid & 31;
    const int lrow = lane & 15;
    const int lhi  = lane >> 4;
    const int row0 = blockIdx.x * BM;
    const int mbase = wave * 16;   // each wave owns 16 batch rows of the tile

    // Init h buffer A with broadcast h0 (bf16).
    for (int idx = tid * 4; idx < BM * HDIM; idx += NW * 32 * 4) {
        const int r = idx >> 8, c = idx & 255;
        const float4 hv = *(const float4*)&h0[c];
        hsA[r * SP + c + 0] = (__bf16)hv.x;
        hsA[r * SP + c + 1] = (__bf16)hv.y;
        hsA[r * SP + c + 2] = (__bf16)hv.z;
        hsA[r * SP + c + 3] = (__bf16)hv.w;
    }

    // Prologue: start DMA of the first weight slice.
#if HAS_TDM
    if (wave == 0) tdm_issue_slice(wih_bf, whh_bf, 0, wsl[0]);
#else
    copy_slice(wih_bf, whh_bf, 0, wsl[0], tid);
#endif

    // x never changes: build this wave's 8 A-fragments (16x32 bf16 each)
    // once, straight from global f32, and keep them in registers (64 VGPRs).
    V16U ax[8];
    {
        const float* xrow = &x[(size_t)(row0 + mbase + lrow) * IDIM];
        #pragma unroll
        for (int kt = 0; kt < 8; ++kt) {
            const int k1 = kt * 32 + lhi * 16;   // A layout: 16 contiguous K at k1
            const float4 f0 = *(const float4*)&xrow[k1 + 0];
            const float4 f1 = *(const float4*)&xrow[k1 + 4];
            const float4 f2 = *(const float4*)&xrow[k1 + 8];
            const float4 f3 = *(const float4*)&xrow[k1 + 12];
            v8bf lo, hi;
            lo[0] = (__bf16)f0.x; lo[1] = (__bf16)f0.y; lo[2] = (__bf16)f0.z; lo[3] = (__bf16)f0.w;
            lo[4] = (__bf16)f1.x; lo[5] = (__bf16)f1.y; lo[6] = (__bf16)f1.z; lo[7] = (__bf16)f1.w;
            hi[0] = (__bf16)f2.x; hi[1] = (__bf16)f2.y; hi[2] = (__bf16)f2.z; hi[3] = (__bf16)f2.w;
            hi[4] = (__bf16)f3.x; hi[5] = (__bf16)f3.y; hi[6] = (__bf16)f3.z; hi[7] = (__bf16)f3.w;
            ax[kt].h[0] = lo; ax[kt].h[1] = hi;
        }
    }
    __syncthreads();

    for (int u = 0; u < UNITS; ++u) {
        const __bf16* hcur = (u & 1) ? hsB : hsA;
        __bf16*       hnxt = (u & 1) ? hsA : hsB;
        const float*  bih  = b_ih + u * G3;
        const float*  bhh  = b_hh + u * G3;

        // h changes once per unit: hoist its 8 A-fragments into registers too.
        V16U ah[8];
        #pragma unroll
        for (int kt = 0; kt < 8; ++kt) {
            const int a0 = (mbase + lrow) * SP + kt * 32 + lhi * 16;
            ah[kt].h[0] = *(const v8bf*)&hcur[a0];
            ah[kt].h[1] = *(const v8bf*)&hcur[a0 + 8];
        }

        #pragma unroll 1
        for (int jt = 0; jt < 16; ++jt) {          // 16 column tiles of H=256
            const int sc = u * 16 + jt;
            const __bf16* wsbuf = wsl[sc & 1];

            // Prefetch next slice into the other buffer; drain this slice's DMA.
#if HAS_TDM
            if (wave == 0) {
                if (sc + 1 < NSLICE) {
                    tdm_issue_slice(wih_bf, whh_bf, sc + 1, wsl[(sc + 1) & 1]);
                    __builtin_amdgcn_s_wait_tensorcnt(6);   // next slice may stay in flight
                } else {
                    __builtin_amdgcn_s_wait_tensorcnt(0);   // final drain
                }
            }
#else
            if (sc + 1 < NSLICE)
                copy_slice(wih_bf, whh_bf, sc + 1, wsl[(sc + 1) & 1], tid);
#endif
            __syncthreads();   // weight slice `sc` visible to all waves

            const int ncol = jt * 16 + lrow;

            // C/D layout: VGPR v, lane l -> M = v + 8*(l>>4), N = l&15.
            v8f a_ir, a_iz, a_in, a_hr, a_hz, a_hn;
            const float bias_ir = bih[ncol], bias_iz = bih[256 + ncol], bias_in = bih[512 + ncol];
            const float bias_hr = bhh[ncol], bias_hz = bhh[256 + ncol], bias_hn = bhh[512 + ncol];
            #pragma unroll
            for (int i = 0; i < 8; ++i) {
                a_ir[i] = bias_ir; a_iz[i] = bias_iz; a_in[i] = bias_in;
                a_hr[i] = bias_hr; a_hz[i] = bias_hz; a_hn[i] = bias_hn;
            }

            #pragma unroll
            for (int kt = 0; kt < 8; ++kt) {       // K = 256 in steps of 32 (full unroll:
                                                   // keeps ax[]/ah[] in VGPRs)
                // B[k][n] = W[n][k]: lane holds 16 contiguous K of column N=lane&15.
                const int brow = lrow * SP + kt * 32 + lhi * 16;
                V16U wb_ir, wb_iz, wb_in, wb_hr, wb_hz, wb_hn;
                wb_ir.h[0] = *(const v8bf*)&wsbuf[          brow];
                wb_ir.h[1] = *(const v8bf*)&wsbuf[          brow + 8];
                wb_iz.h[0] = *(const v8bf*)&wsbuf[16 * SP + brow];
                wb_iz.h[1] = *(const v8bf*)&wsbuf[16 * SP + brow + 8];
                wb_in.h[0] = *(const v8bf*)&wsbuf[32 * SP + brow];
                wb_in.h[1] = *(const v8bf*)&wsbuf[32 * SP + brow + 8];
                wb_hr.h[0] = *(const v8bf*)&wsbuf[48 * SP + brow];
                wb_hr.h[1] = *(const v8bf*)&wsbuf[48 * SP + brow + 8];
                wb_hz.h[0] = *(const v8bf*)&wsbuf[64 * SP + brow];
                wb_hz.h[1] = *(const v8bf*)&wsbuf[64 * SP + brow + 8];
                wb_hn.h[0] = *(const v8bf*)&wsbuf[80 * SP + brow];
                wb_hn.h[1] = *(const v8bf*)&wsbuf[80 * SP + brow + 8];

                a_ir = __builtin_amdgcn_wmma_f32_16x16x32_bf16(false, ax[kt].v, false, wb_ir.v, (short)0, a_ir, false, false);
                a_iz = __builtin_amdgcn_wmma_f32_16x16x32_bf16(false, ax[kt].v, false, wb_iz.v, (short)0, a_iz, false, false);
                a_in = __builtin_amdgcn_wmma_f32_16x16x32_bf16(false, ax[kt].v, false, wb_in.v, (short)0, a_in, false, false);
                a_hr = __builtin_amdgcn_wmma_f32_16x16x32_bf16(false, ah[kt].v, false, wb_hr.v, (short)0, a_hr, false, false);
                a_hz = __builtin_amdgcn_wmma_f32_16x16x32_bf16(false, ah[kt].v, false, wb_hz.v, (short)0, a_hz, false, false);
                a_hn = __builtin_amdgcn_wmma_f32_16x16x32_bf16(false, ah[kt].v, false, wb_hn.v, (short)0, a_hn, false, false);
            }

            // Gate math in f32; write h_new (bf16) to the other LDS buffer,
            // or stream f32 to global on the last unit.
            #pragma unroll
            for (int v = 0; v < 8; ++v) {
                const int row = mbase + v + lhi * 8;
                const float r    = fsigmoid(a_ir[v] + a_hr[v]);
                const float z    = fsigmoid(a_iz[v] + a_hz[v]);
                const float nn   = ftanh_fast(a_in[v] + r * a_hn[v]);
                const float hold = (float)hcur[row * SP + ncol];
                const float hnew = (1.0f - z) * nn + z * hold;
                if (u == UNITS - 1)
                    out[(size_t)(row0 + row) * HDIM + ncol] = hnew;
                else
                    hnxt[row * SP + ncol] = (__bf16)hnew;
            }
            __syncthreads();   // all waves done with slice sc before its buffer is reused
        }
    }
}

extern "C" void kernel_launch(void* const* d_in, const int* in_sizes, int n_in,
                              void* d_out, int out_size, void* d_ws, size_t ws_size,
                              hipStream_t stream) {
    const float* x   = (const float*)d_in[0];
    const float* Wih = (const float*)d_in[1];
    const float* Whh = (const float*)d_in[2];
    const float* bih = (const float*)d_in[3];
    const float* bhh = (const float*)d_in[4];
    const float* h0  = (const float*)d_in[5];
    float* out = (float*)d_out;

    const int wN = UNITS * G3 * IDIM;          // 1,572,864 elems per weight tensor
    __bf16* wih_bf = (__bf16*)d_ws;            // 3 MB
    __bf16* whh_bf = wih_bf + wN;              // +3 MB (ws usage ~6.3 MB)

    cvt_bf16_kernel<<<(wN + 255) / 256, 256, 0, stream>>>(Wih, wih_bf, wN);
    cvt_bf16_kernel<<<(wN + 255) / 256, 256, 0, stream>>>(Whh, whh_bf, wN);

    gru_chain_kernel<<<BATCH / BM, NW * 32, 0, stream>>>(
        x, wih_bf, whh_bf, bih, bhh, h0, out);
}
